// MultiLayerRangeAttentionV2_8108898255323
// MI455X (gfx1250) — matrix operand order repaired and checked by reading, not compiled, not run on hardware
//
#include <hip/hip_runtime.h>

typedef __attribute__((ext_vector_type(2)))  float    v2f;
typedef __attribute__((ext_vector_type(8)))  float    v8f;
typedef __attribute__((ext_vector_type(8)))  _Float16 v8h;
typedef __attribute__((ext_vector_type(16))) _Float16 v16h;

#define RNG    512
#define KTOK   250
#define KPAD   256
#define EDIM   16
#define NLAYER 2
#define LN_EPS 1e-5f

// xor-shuffle within 16-lane groups via ds_swizzle immediate (no address VALU):
// group-of-32 encoding: offset[14:10]=xor_mask, offset[9:5]=or_mask, offset[4:0]=and_mask
template <int MASK>
__device__ __forceinline__ float swz_xor_f(float v) {
  union { float f; int i; } u;
  u.f = v;
  u.i = __builtin_amdgcn_ds_swizzle(u.i, (MASK << 10) | 0x1f);
  return u.f;
}
__device__ __forceinline__ float rmax16(float v) {
  v = fmaxf(v, swz_xor_f<1>(v));
  v = fmaxf(v, swz_xor_f<2>(v));
  v = fmaxf(v, swz_xor_f<4>(v));
  v = fmaxf(v, swz_xor_f<8>(v));
  return v;
}
__device__ __forceinline__ float rsum16(float v) {
  v += swz_xor_f<1>(v);
  v += swz_xor_f<2>(v);
  v += swz_xor_f<4>(v);
  v += swz_xor_f<8>(v);
  return v;
}

__global__ __launch_bounds__(256) void mlra_kernel(
    const float* __restrict__ power_vals,
    const int*   __restrict__ ele_idx,
    const int*   __restrict__ azi_idx,
    const float* __restrict__ vp_w,    // [8][8]
    const float* __restrict__ vp_b,    // [8]
    const float* __restrict__ ele_emb, // [37][4]
    const float* __restrict__ azi_emb, // [107][4]
    const float* __restrict__ in_w,    // [L][48][16]
    const float* __restrict__ in_b,    // [L][48]
    const float* __restrict__ out_w,   // [L][16][16]
    const float* __restrict__ out_b,   // [L][16]
    const float* __restrict__ ln_g,    // [16]
    const float* __restrict__ ln_b,    // [16]
    float* __restrict__ out)           // [R*K][16]
{
  // Per-range resident state (one workgroup per range). ~92 KB of the WGP's 320 KB LDS.
  __shared__ __align__(16) float    xs[KPAD][EDIM];       // activations (residual stream)
  __shared__ __align__(16) float    qs[KPAD][EDIM];       // Q (pre-scaled by 1/sqrt(D))
  __shared__ __align__(16) float    ksm[KPAD][EDIM];      // K
  __shared__ __align__(16) _Float16 vsm[KPAD][EDIM];      // V in f16 (feeds f16 WMMA B)
  __shared__ __align__(16) float    os[KPAD][EDIM];       // attention output
  __shared__ __align__(16) _Float16 pbuf[8][16][64];      // per-wave P transpose scratch
  __shared__ __align__(16) float    w_in_s[48][EDIM];
  __shared__               float    b_in_s[48];
  __shared__ __align__(16) float    w_out_s[EDIM][EDIM];
  __shared__               float    b_out_s[EDIM];
  __shared__               float    lng_s[EDIM], lnb_s[EDIM];

  const int r    = blockIdx.x;
  const int tid  = threadIdx.x;
  const int lane = tid & 31;
  const int half = lane >> 4;   // which 16-lane half of the wave
  const int l15  = lane & 15;
  // wave index is uniform across the wave: force it into an SGPR so all task
  // decomposition (ob/tt/qt/h) and branches become scalar control flow.
  const int wave = __builtin_amdgcn_readfirstlane(tid >> 5);

  // ---------------- stage 0: x = [pv @ vp_w^T + vp_b, ele_emb, azi_emb] ----------------
  {
    const int t = tid;
    float row[EDIM];
    if (t < KTOK) {
      const float* pv = power_vals + (size_t)(r * KTOK + t) * 8;
      float p[8];
      #pragma unroll
      for (int i = 0; i < 8; ++i) p[i] = pv[i];
      #pragma unroll
      for (int j = 0; j < 8; ++j) {
        float a = vp_b[j];
        #pragma unroll
        for (int i = 0; i < 8; ++i) a += p[i] * vp_w[j * 8 + i];
        row[j] = a;
      }
      const int ei = ele_idx[r * KTOK + t];
      const int ai = azi_idx[r * KTOK + t];
      #pragma unroll
      for (int d = 0; d < 4; ++d) {
        row[8 + d]  = ele_emb[ei * 4 + d];
        row[12 + d] = azi_emb[ai * 4 + d];
      }
    } else {
      #pragma unroll
      for (int j = 0; j < EDIM; ++j) row[j] = 0.f;  // pad tokens 250..255 stay zero
    }
    #pragma unroll
    for (int j = 0; j < EDIM; ++j) xs[t][j] = row[j];
  }
  if (tid < EDIM) { lng_s[tid] = ln_g[tid]; lnb_s[tid] = ln_b[tid]; }
  __syncthreads();

  for (int l = 0; l < NLAYER; ++l) {
    // ---------------- layer weights -> LDS ----------------
    {
      const float* iw = in_w + l * 48 * EDIM;
      for (int i = tid; i < 48 * EDIM; i += 256) w_in_s[i / EDIM][i % EDIM] = iw[i];
      const float* ow = out_w + l * EDIM * EDIM;
      if (tid < EDIM * EDIM) w_out_s[tid / EDIM][tid % EDIM] = ow[tid];
      if (tid < 48)   b_in_s[tid]  = in_b[l * 48 + tid];
      if (tid < EDIM) b_out_s[tid] = out_b[l * EDIM + tid];
    }
    __syncthreads();

    // ---------------- QKV projection: WMMA f32 16x16x4, contract E=16 in 4 chunks ----------------
    // 48 groups = 16 token-tiles x 3 output blocks (q,k,v); g is scalar -> scalar branches.
    for (int g = wave; g < 48; g += 8) {
      const int tt   = g & 15;      // token tile          (SGPR)
      const int ob   = g >> 4;      // 0=q, 1=k, 2=v       (SGPR)
      const int kofs = half ? 2 : 0;
      v8f acc;
      const float bias = b_in_s[ob * EDIM + l15];
      #pragma unroll
      for (int i = 0; i < 8; ++i) acc[i] = bias;  // C layout: lane = out column -> bias broadcast
      #pragma unroll
      for (int ic = 0; ic < 4; ++ic) {
        // A[m][k]: lane = m, VGPR pair = k within 4-chunk (lane half selects k 0,1 vs 2,3)
        v2f a = *(const v2f*)&xs[tt * 16 + l15][ic * 4 + kofs];
        // B[k][n] = W[n][k]: lane = n column
        v2f b = *(const v2f*)&w_in_s[ob * EDIM + l15][ic * 4 + kofs];
        acc = __builtin_amdgcn_wmma_f32_16x16x4_f32(false, a, false, b, (short)0, acc, false, false);
      }
      if (ob == 0) {                 // scalar branch
        #pragma unroll
        for (int g8 = 0; g8 < 8; ++g8)
          qs[tt * 16 + g8 + 8 * half][l15] = acc[g8] * 0.5f;   // fold scale = 1/sqrt(4)
      } else if (ob == 1) {
        #pragma unroll
        for (int g8 = 0; g8 < 8; ++g8)
          ksm[tt * 16 + g8 + 8 * half][l15] = acc[g8];
      } else {
        // Pad rows (250..255) need only be finite: their attention weights are
        // exactly 0 after the -1e30 mask, so no predicate needed here.
        #pragma unroll
        for (int g8 = 0; g8 < 8; ++g8)
          vsm[tt * 16 + g8 + 8 * half][l15] = (_Float16)acc[g8];
      }
    }
    __syncthreads();

    // ---------------- attention: 64 tasks = 16 q-tiles x 4 heads ----------------
    for (int task = wave; task < 64; task += 8) {
      const int qt   = task & 15;   // SGPR
      const int h    = task >> 4;   // SGPR
      const int kofs = half ? 2 : 0;
      // Q tile for this head: A of the 16x16x4 f32 WMMA (reused for all key tiles)
      const v2f aq = *(const v2f*)&qs[qt * 16 + l15][4 * h + kofs];

      float m[8], s[8];
      v8f O;
      #pragma unroll
      for (int i = 0; i < 8; ++i) { m[i] = -1e30f; s[i] = 0.f; O[i] = 0.f; }

      for (int kc = 0; kc < 4; ++kc) {            // 4 chunks of 64 keys (250 padded to 256)
        v8f att[4];
        #pragma unroll
        for (int j = 0; j < 4; ++j) {             // QK^T: one f32 WMMA per 16x16 tile
          const int kt = kc * 4 + j;
          v2f bk = *(const v2f*)&ksm[kt * 16 + l15][4 * h + kofs];
          v8f z;
          #pragma unroll
          for (int i = 0; i < 8; ++i) z[i] = 0.f;
          att[j] = __builtin_amdgcn_wmma_f32_16x16x4_f32(false, aq, false, bk, (short)0, z, false, false);
        }
        if (kc == 3 && l15 >= 10) {               // mask pad keys 250..255 (lane = key column)
          #pragma unroll
          for (int i = 0; i < 8; ++i) att[3][i] = -1e30f;
        }
        // online softmax: rows live in VGPR index (lane-local), columns reduced via ds_swizzle
        #pragma unroll
        for (int g8 = 0; g8 < 8; ++g8) {
          float cm = fmaxf(fmaxf(att[0][g8], att[1][g8]), fmaxf(att[2][g8], att[3][g8]));
          cm = rmax16(cm);
          const float mn    = fmaxf(m[g8], cm);
          const float alpha = __expf(m[g8] - mn);
          float rs = 0.f;
          #pragma unroll
          for (int j = 0; j < 4; ++j) {
            const float p = __expf(att[j][g8] - mn);
            att[j][g8] = p;
            rs += p;
          }
          rs    = rsum16(rs);
          s[g8] = s[g8] * alpha + rs;
          m[g8] = mn;
          O[g8] *= alpha;
        }
        // transpose P (D layout: lane=col) -> pbuf rows (A layout needs lane=row), as f16
        #pragma unroll
        for (int j = 0; j < 4; ++j)
          #pragma unroll
          for (int g8 = 0; g8 < 8; ++g8)
            pbuf[wave][g8 + 8 * half][j * 16 + l15] = (_Float16)att[j][g8];
        asm volatile("s_wait_dscnt 0" ::: "memory");  // CDNA5 split counter: stores -> loads

        // PV: f16 WMMA 16x16x32, B = V rows for all 4 heads (extract our 4 cols at the end)
        #pragma unroll
        for (int kk = 0; kk < 2; ++kk) {
          const int koff = half ? 8 : 0;          // A layout: lane half interleaves K groups of 8
          v8h alo = *(const v8h*)&pbuf[wave][l15][kk * 32 + koff];
          v8h ahi = *(const v8h*)&pbuf[wave][l15][kk * 32 + 16 + koff];
          v16h A = __builtin_shufflevector(alo, ahi, 0,1,2,3,4,5,6,7,8,9,10,11,12,13,14,15);
          const int tok = kc * 64 + kk * 32 + koff;  // B layout: lane = V column, elements = key rows (same K interleave as A)
          v16h B;
          #pragma unroll
          for (int s2 = 0; s2 < 16; ++s2) B[s2] = vsm[tok + s2 + (s2 >= 8 ? 8 : 0)][l15];
          O = __builtin_amdgcn_wmma_f32_16x16x32_f16(false, A, false, B, (short)0, O, false, false);
        }
      }
      // write this head's 4 output columns (lane = column)
      if ((l15 >> 2) == h) {
        #pragma unroll
        for (int g8 = 0; g8 < 8; ++g8)
          os[qt * 16 + g8 + 8 * half][l15] = O[g8] / s[g8];
      }
    }
    __syncthreads();

    // ---------------- out-proj + residual + shared LayerNorm (one token per thread) ----------------
    {
      const int t = tid;
      float o[EDIM], y[EDIM];
      #pragma unroll
      for (int i = 0; i < EDIM; ++i) o[i] = os[t][i];
      float mu = 0.f;
      #pragma unroll
      for (int j = 0; j < EDIM; ++j) {
        float a = b_out_s[j] + xs[t][j];
        #pragma unroll
        for (int i = 0; i < EDIM; ++i) a += o[i] * w_out_s[j][i];
        y[j] = a;
        mu += a;
      }
      mu *= (1.f / EDIM);
      float var = 0.f;
      #pragma unroll
      for (int j = 0; j < EDIM; ++j) { const float d = y[j] - mu; var += d * d; }
      var *= (1.f / EDIM);
      const float rstd = rsqrtf(var + LN_EPS);
      #pragma unroll
      for (int j = 0; j < EDIM; ++j) {
        const float res = (y[j] - mu) * rstd * lng_s[j] + lnb_s[j];
        xs[t][j] = (t < KTOK) ? res : 0.f;   // keep pad rows zeroed for next layer
        if (l == NLAYER - 1 && t < KTOK)
          out[(size_t)(r * KTOK + t) * EDIM + j] = res;
      }
    }
    __syncthreads();
  }
}

extern "C" void kernel_launch(void* const* d_in, const int* in_sizes, int n_in,
                              void* d_out, int out_size, void* d_ws, size_t ws_size,
                              hipStream_t stream) {
  (void)in_sizes; (void)n_in; (void)out_size; (void)d_ws; (void)ws_size;
  const float* power_vals = (const float*)d_in[0];
  const int*   ele_idx    = (const int*)  d_in[1];
  const int*   azi_idx    = (const int*)  d_in[2];
  const float* vp_w       = (const float*)d_in[3];
  const float* vp_b       = (const float*)d_in[4];
  const float* ele_emb    = (const float*)d_in[5];
  const float* azi_emb    = (const float*)d_in[6];
  const float* in_w       = (const float*)d_in[7];
  const float* in_b       = (const float*)d_in[8];
  const float* out_w      = (const float*)d_in[9];
  const float* out_b      = (const float*)d_in[10];
  const float* ln_g       = (const float*)d_in[11];
  const float* ln_b       = (const float*)d_in[12];
  float*       out        = (float*)d_out;

  mlra_kernel<<<RNG, 256, 0, stream>>>(power_vals, ele_idx, azi_idx, vp_w, vp_b,
                                       ele_emb, azi_emb, in_w, in_b, out_w, out_b,
                                       ln_g, ln_b, out);
}
